// Linformer_15401752724220
// MI455X (gfx1250) — matrix operand order repaired and checked by reading, not compiled, not run on hardware
//
#include <hip/hip_runtime.h>
#include <math.h>

// ---------------------------------------------------------------------------
// Feature detection (compile-safe on both toolchains / host+device passes)
// ---------------------------------------------------------------------------
#if defined(__has_builtin)
#  if __has_builtin(__builtin_amdgcn_global_load_async_to_lds_b128)
#    define HAVE_ASYNC_LDS 1
#  endif
#  if __has_builtin(__builtin_amdgcn_s_wait_asynccnt)
#    define HAVE_WAIT_ASYNC_BUILTIN 1
#  endif
#endif
#ifndef HAVE_ASYNC_LDS
#  define HAVE_ASYNC_LDS 0
#endif
#ifndef HAVE_WAIT_ASYNC_BUILTIN
#  define HAVE_WAIT_ASYNC_BUILTIN 0
#endif

#if HAVE_ASYNC_LDS
#  if HAVE_WAIT_ASYNC_BUILTIN
#    define WAIT_ASYNC() __builtin_amdgcn_s_wait_asynccnt(0)
#  else
#    define WAIT_ASYNC() asm volatile("s_wait_asynccnt 0x0" ::: "memory")
#  endif
#else
#  define WAIT_ASYNC() ((void)0)
#endif

#define AS1 __attribute__((address_space(1)))
#define AS3 __attribute__((address_space(3)))

// ---------------------------------------------------------------------------
// Types
// ---------------------------------------------------------------------------
typedef __attribute__((ext_vector_type(16))) __bf16       v16bf;
typedef __attribute__((ext_vector_type(8)))  float        v8f;
typedef __attribute__((ext_vector_type(4)))  unsigned int v4u;
typedef __attribute__((ext_vector_type(4)))  int          v4i;

// 16-byte views for the async-DMA builtin (wants v4i* in AS1 / AS3)
__device__ __forceinline__ AS1 v4i* glb16(const unsigned short* p) {
    return (AS1 v4i*)(AS1 unsigned short*)const_cast<unsigned short*>(p);
}
__device__ __forceinline__ AS3 v4i* lds16(unsigned short* p) {
    return (AS3 v4i*)(AS3 unsigned short*)p;
}

union FragB16 {           // 32 bytes: one WMMA bf16 A/B operand per lane
    v16bf v;
    v4u   q[2];
};

__device__ __forceinline__ unsigned short f2bf(float f) {
    unsigned int u = __float_as_uint(f);
    unsigned int r = u + 0x7FFFu + ((u >> 16) & 1u);   // round-to-nearest-even
    return (unsigned short)(r >> 16);
}

// ---------------------------------------------------------------------------
// Tiled batched WMMA GEMM:  C[M,N] = alpha * A[M,K] x op(B) (+ bias) (+ act)
//   A: bf16 row-major [M,K] stride lda          (per-batch offset sA1/sA2)
//   B: bf16;  BT=true : row-major [N,K] stride ldb  (both operands K-contig)
//             BT=false: row-major [K,N] stride ldb  (transposed into LDS)
//   C: f32 and/or bf16, stride ldc
//   biasMode: 0 none, 1 bias[col], 2 bias[row]   act: 0 none, 1 exact GeLU
//
// Block: 256 threads (8 waves). Tile 128x128x32. Wave tile 32x64 = 2x4 WMMA.
// Double-buffered LDS pipeline. Tiles are staged with the gfx1250 async DMA
// (GLOBAL_LOAD_ASYNC_TO_LDS_B128, tracked by ASYNCcnt) so the matrix pipe
// never sees the global->VGPR->LDS round trip; a manual load+ds_store
// pipeline with identical buffering is kept as fallback.
//
// REQUIREMENTS (all satisfied by every call below):
//   M % 128 == 0, K % 32 == 0; BT=true additionally requires N % 128 == 0.
// ---------------------------------------------------------------------------
#define TBM 128
#define TBN 128
#define TBK 32

template <bool BT>
__global__ __launch_bounds__(256) void wmma_gemm(
    const unsigned short* __restrict__ A,
    const unsigned short* __restrict__ Bm,
    float* __restrict__ Cf,
    unsigned short* __restrict__ Ch,
    const float* __restrict__ bias,
    int M, int N, int K, int lda, int ldb, int ldc, int batch2,
    long sA1, long sA2, long sB1, long sB2, long sC1, long sC2,
    float alpha, int biasMode, int act)
{
    __shared__ unsigned short As[2][TBM * TBK];   // [buf][row][k]
    __shared__ unsigned short Bs[2][TBN * TBK];   // [buf][col][k]

    const int z  = blockIdx.z;
    const int i1 = z / batch2;
    const int i2 = z - i1 * batch2;
    A  += (size_t)(i1 * sA1 + i2 * sA2);
    Bm += (size_t)(i1 * sB1 + i2 * sB2);
    const long coff = i1 * sC1 + i2 * sC2;

    const int tileM = blockIdx.y * TBM;
    const int tileN = blockIdx.x * TBN;

    const int tid  = threadIdx.x;
    const int lane = tid & 31;
    const int wave = tid >> 5;
    const int wm   = wave >> 1;      // 0..3 : 32-row band
    const int wn   = wave & 1;       // 0..1 : 64-col band
    const int hl   = lane >> 4;      // lane half (WMMA layout)
    const int l16  = lane & 15;

    // ---- per-thread staging coordinates (guard-free) ---------------------
    const int arow = tid >> 2;            // 0..63
    const int acol = (tid & 3) * 8;       // 0,8,16,24
    const int asO0 = arow * TBK + acol;
    const int asO1 = (arow + 64) * TBK + acol;
    const unsigned short* aP0 = A + (size_t)(tileM + arow) * lda + acol;
    const unsigned short* aP1 = aP0 + (size_t)64 * lda;

    const unsigned short* bP0;
    const unsigned short* bP1;
    int  bk = 0, bn = 0;
    bool bIn = true;
    if (BT) {
        bP0 = Bm + (size_t)(tileN + arow) * ldb + acol;
        bP1 = bP0 + (size_t)64 * ldb;
    } else {
        bk  = tid >> 4;                   // 0..15 (k row; second pass +16)
        bn  = (tid & 15) * 8;             // 0..120
        bP0 = Bm + (size_t)bk * ldb + tileN + bn;
        bP1 = bP0 + (size_t)16 * ldb;
        bIn = (tileN + bn + 8 <= N);
    }

    // ---- accumulators ----------------------------------------------------
    v8f acc[2][4];
#pragma unroll
    for (int i = 0; i < 2; ++i)
#pragma unroll
        for (int j = 0; j < 4; ++j)
#pragma unroll
            for (int e = 0; e < 8; ++e) acc[i][j][e] = 0.0f;

    const int nk = K / TBK;

    // ---- prologue: stage tile 0 into buffer 0 ----------------------------
    {
#if HAVE_ASYNC_LDS
        __builtin_amdgcn_global_load_async_to_lds_b128(
            glb16(aP0), lds16(&As[0][asO0]), 0, 0);
        __builtin_amdgcn_global_load_async_to_lds_b128(
            glb16(aP1), lds16(&As[0][asO1]), 0, 0);
        if (BT) {
            __builtin_amdgcn_global_load_async_to_lds_b128(
                glb16(bP0), lds16(&Bs[0][asO0]), 0, 0);
            __builtin_amdgcn_global_load_async_to_lds_b128(
                glb16(bP1), lds16(&Bs[0][asO1]), 0, 0);
        }
#else
        {
            v4u a0 = *(const v4u*)aP0;
            v4u a1 = *(const v4u*)aP1;
            *(v4u*)(&As[0][asO0]) = a0;
            *(v4u*)(&As[0][asO1]) = a1;
            if (BT) {
                v4u b0 = *(const v4u*)bP0;
                v4u b1 = *(const v4u*)bP1;
                *(v4u*)(&Bs[0][asO0]) = b0;
                *(v4u*)(&Bs[0][asO1]) = b1;
            }
        }
#endif
        if (!BT) {
            v4u b0 = {0u, 0u, 0u, 0u}, b1 = {0u, 0u, 0u, 0u};
            if (bIn) {
                b0 = *(const v4u*)bP0;
                b1 = *(const v4u*)bP1;
            }
#pragma unroll
            for (int e = 0; e < 8; ++e) {
                Bs[0][(bn + e) * TBK + bk] =
                    (unsigned short)(b0[e >> 1] >> ((e & 1) * 16));
                Bs[0][(bn + e) * TBK + bk + 16] =
                    (unsigned short)(b1[e >> 1] >> ((e & 1) * 16));
            }
        }
        aP0 += TBK;
        aP1 += TBK;
        if (BT) { bP0 += TBK; bP1 += TBK; }
        else    { bP0 += (size_t)TBK * ldb; bP1 += (size_t)TBK * ldb; }
    }
    WAIT_ASYNC();
    __syncthreads();

    // ---- pipelined main loop ---------------------------------------------
    const int aF = (wm * 32 + l16) * TBK;      // A frag LDS base (per lane)
    const int bF = (wn * 64 + l16) * TBK;      // B frag LDS base (per lane)

    for (int kb = 0; kb < nk; ++kb) {
        const int  cur  = kb & 1;
        const int  nxt  = cur ^ 1;
        const bool more = (kb + 1) < nk;

        // issue staging for next tile (in flight behind the WMMAs)
        v4u b0 = {0u, 0u, 0u, 0u}, b1 = {0u, 0u, 0u, 0u};  // transpose path
#if !HAVE_ASYNC_LDS
        v4u a0 = {0u, 0u, 0u, 0u}, a1 = {0u, 0u, 0u, 0u};
        v4u c0 = {0u, 0u, 0u, 0u}, c1 = {0u, 0u, 0u, 0u};  // BT B staging
#endif
        if (more) {
#if HAVE_ASYNC_LDS
            __builtin_amdgcn_global_load_async_to_lds_b128(
                glb16(aP0), lds16(&As[nxt][asO0]), 0, 0);
            __builtin_amdgcn_global_load_async_to_lds_b128(
                glb16(aP1), lds16(&As[nxt][asO1]), 0, 0);
            if (BT) {
                __builtin_amdgcn_global_load_async_to_lds_b128(
                    glb16(bP0), lds16(&Bs[nxt][asO0]), 0, 0);
                __builtin_amdgcn_global_load_async_to_lds_b128(
                    glb16(bP1), lds16(&Bs[nxt][asO1]), 0, 0);
            }
#else
            a0 = *(const v4u*)aP0;
            a1 = *(const v4u*)aP1;
            if (BT) {
                c0 = *(const v4u*)bP0;
                c1 = *(const v4u*)bP1;
            }
#endif
            if (!BT && bIn) {
                b0 = *(const v4u*)bP0;
                b1 = *(const v4u*)bP1;
            }
            aP0 += TBK;
            aP1 += TBK;
            if (BT) { bP0 += TBK; bP1 += TBK; }
            else    { bP0 += (size_t)TBK * ldb; bP1 += (size_t)TBK * ldb; }
        }

        // ---- gather WMMA fragments from current buffer (ds_load_b128) ----
        // A 16-bit 16x32 layout: lanes 0-15 hold K 0-7 & 16-23, lanes 16-31
        // hold K 8-15 & 24-31 (row = lane&15).
        FragB16 af[2];
#pragma unroll
        for (int tm = 0; tm < 2; ++tm) {
            const unsigned short* ap = &As[cur][aF + tm * 16 * TBK];
            af[tm].q[0] = *(const v4u*)(ap + 8 * hl);
            af[tm].q[1] = *(const v4u*)(ap + 16 + 8 * hl);
        }
        // B 16-bit 32x16 layout: lanes 0-15 hold K 0-7 & 16-23, lanes 16-31 K 8-15 & 24-31
        FragB16 bfr[4];
#pragma unroll
        for (int tn = 0; tn < 4; ++tn) {
            const unsigned short* bp = &Bs[cur][bF + tn * 16 * TBK];
            bfr[tn].q[0] = *(const v4u*)(bp + 8 * hl);
            bfr[tn].q[1] = *(const v4u*)(bp + 16 + 8 * hl);
        }

#pragma unroll
        for (int tm = 0; tm < 2; ++tm)
#pragma unroll
            for (int tn = 0; tn < 4; ++tn)
                acc[tm][tn] = __builtin_amdgcn_wmma_f32_16x16x32_bf16(
                    false, af[tm].v, false, bfr[tn].v,
                    (short)0, acc[tm][tn], false, false);

        // ---- commit next tile into the other buffer ----------------------
        if (more) {
#if !HAVE_ASYNC_LDS
            *(v4u*)(&As[nxt][asO0]) = a0;
            *(v4u*)(&As[nxt][asO1]) = a1;
            if (BT) {
                *(v4u*)(&Bs[nxt][asO0]) = c0;
                *(v4u*)(&Bs[nxt][asO1]) = c1;
            }
#endif
            if (!BT) {
#pragma unroll
                for (int e = 0; e < 8; ++e) {
                    Bs[nxt][(bn + e) * TBK + bk] =
                        (unsigned short)(b0[e >> 1] >> ((e & 1) * 16));
                    Bs[nxt][(bn + e) * TBK + bk + 16] =
                        (unsigned short)(b1[e >> 1] >> ((e & 1) * 16));
                }
            }
        }
        WAIT_ASYNC();
        __syncthreads();
    }

    // ---- epilogue: alpha, bias, activation, f32 / bf16 stores ------------
    float* cf          = Cf ? Cf + coff : nullptr;
    unsigned short* ch = Ch ? Ch + coff : nullptr;
#pragma unroll
    for (int tm = 0; tm < 2; ++tm) {
#pragma unroll
        for (int tn = 0; tn < 4; ++tn) {
            const int col = tileN + wn * 64 + tn * 16 + l16;
            if (col >= N) continue;
#pragma unroll
            for (int e = 0; e < 8; ++e) {
                const int row = tileM + wm * 32 + tm * 16 + 8 * hl + e;
                float v = acc[tm][tn][e] * alpha;
                if (biasMode == 1)      v += bias[col];
                else if (biasMode == 2) v += bias[row];
                if (act == 1)
                    v = 0.5f * v * (1.0f + erff(v * 0.70710678118654752f));
                const size_t o = (size_t)row * ldc + col;
                if (cf) cf[o] = v;
                if (ch) ch[o] = f2bf(v);
            }
        }
    }
}

// ---------------------------------------------------------------------------
// Softmax over rows of length 256 (k proj dim); scale folded in; bf16 out.
// ---------------------------------------------------------------------------
__global__ __launch_bounds__(256) void softmax256(
    const float* __restrict__ S, unsigned short* __restrict__ P, float scale)
{
    const size_t row = blockIdx.x;
    const int t = threadIdx.x;
    const float* s   = S + row * 256;
    unsigned short* p = P + row * 256;
    __shared__ float red[256];
    float v = s[t] * scale;
    red[t] = v;
    __syncthreads();
    for (int o = 128; o > 0; o >>= 1) {
        if (t < o) red[t] = fmaxf(red[t], red[t + o]);
        __syncthreads();
    }
    const float m = red[0];
    __syncthreads();
    const float e = __expf(v - m);
    red[t] = e;
    __syncthreads();
    for (int o = 128; o > 0; o >>= 1) {
        if (t < o) red[t] += red[t + o];
        __syncthreads();
    }
    p[t] = f2bf(e * (1.0f / red[0]));
}

// ---------------------------------------------------------------------------
// x_out = LN(x_in + delta) over C=512; also emits bf16 copy for next GEMM.
// ---------------------------------------------------------------------------
__global__ __launch_bounds__(256) void residual_ln512(
    const float* __restrict__ x_in, const float* __restrict__ delta,
    const float* __restrict__ g, const float* __restrict__ b,
    float* __restrict__ x_out, unsigned short* __restrict__ xb_out)
{
    const size_t row = blockIdx.x;
    const int t = threadIdx.x;           // 256 threads, 2 elems each
    const float* xi = x_in + row * 512;
    const float* dl = delta + row * 512;
    const float v0 = xi[t] + dl[t];
    const float v1 = xi[t + 256] + dl[t + 256];
    __shared__ float red[256];
    red[t] = v0 + v1;
    __syncthreads();
    for (int o = 128; o > 0; o >>= 1) {
        if (t < o) red[t] += red[t + o];
        __syncthreads();
    }
    const float mean = red[0] * (1.0f / 512.0f);
    __syncthreads();
    const float d0 = v0 - mean, d1 = v1 - mean;
    red[t] = d0 * d0 + d1 * d1;
    __syncthreads();
    for (int o = 128; o > 0; o >>= 1) {
        if (t < o) red[t] += red[t + o];
        __syncthreads();
    }
    const float rstd = rsqrtf(red[0] * (1.0f / 512.0f) + 1e-5f);
    const float y0 = d0 * rstd * g[t] + b[t];
    const float y1 = d1 * rstd * g[t + 256] + b[t + 256];
    float* xo = x_out + row * 512;
    xo[t] = y0;
    xo[t + 256] = y1;
    if (xb_out) {
        unsigned short* xb = xb_out + row * 512;
        xb[t] = f2bf(y0);
        xb[t + 256] = f2bf(y1);
    }
}

// ---------------------------------------------------------------------------
// f32 -> bf16 conversion; and f32 -> (f32 copy + bf16).
// ---------------------------------------------------------------------------
__global__ void cvt_bf16(const float* __restrict__ in,
                         unsigned short* __restrict__ out, long n)
{
    const long i = (long)blockIdx.x * blockDim.x + threadIdx.x;
    if (i < n) out[i] = f2bf(in[i]);
}

__global__ void copy_cvt(const float* __restrict__ in,
                         float* __restrict__ outf,
                         unsigned short* __restrict__ outh, long n)
{
    const long i = (long)blockIdx.x * blockDim.x + threadIdx.x;
    if (i < n) {
        const float v = in[i];
        outf[i] = v;
        outh[i] = f2bf(v);
    }
}

// ---------------------------------------------------------------------------
// Host orchestration
// ---------------------------------------------------------------------------
static void launch_gemm(hipStream_t s, bool bt,
                        const unsigned short* A, const unsigned short* B,
                        float* Cf, unsigned short* Ch, const float* bias,
                        int M, int N, int K, int lda, int ldb, int ldc,
                        int batches, int batch2,
                        long sA1, long sA2, long sB1, long sB2,
                        long sC1, long sC2,
                        float alpha, int biasMode, int act)
{
    dim3 grid((N + TBN - 1) / TBN, (M + TBM - 1) / TBM, batches);
    dim3 blk(256);
    if (bt)
        wmma_gemm<true><<<grid, blk, 0, s>>>(A, B, Cf, Ch, bias, M, N, K,
            lda, ldb, ldc, batch2, sA1, sA2, sB1, sB2, sC1, sC2,
            alpha, biasMode, act);
    else
        wmma_gemm<false><<<grid, blk, 0, s>>>(A, B, Cf, Ch, bias, M, N, K,
            lda, ldb, ldc, batch2, sA1, sA2, sB1, sB2, sC1, sC2,
            alpha, biasMode, act);
}

extern "C" void kernel_launch(void* const* d_in, const int* in_sizes, int n_in,
                              void* d_out, int out_size, void* d_ws, size_t ws_size,
                              hipStream_t stream)
{
    (void)in_sizes; (void)n_in; (void)out_size; (void)ws_size;

    constexpr int Bb = 4, Ns = 4096, Cd = 512, Hh = 8, Dh = 64;
    constexpr int Kdp = 256, Ffd = 2048, Ll = 2;
    constexpr int Tok = Bb * Ns;                 // 16384 tokens

    const float* x     = (const float*)d_in[0];
    const float* Wq    = (const float*)d_in[1];
    const float* Wk    = (const float*)d_in[2];
    const float* Wv    = (const float*)d_in[3];
    const float* Wo    = (const float*)d_in[4];
    const float* bo    = (const float*)d_in[5];
    const float* ln1_g = (const float*)d_in[6];
    const float* ln1_b = (const float*)d_in[7];
    const float* W1    = (const float*)d_in[8];
    const float* b1    = (const float*)d_in[9];
    const float* W2    = (const float*)d_in[10];
    const float* b2    = (const float*)d_in[11];
    const float* ln2_g = (const float*)d_in[12];
    const float* ln2_b = (const float*)d_in[13];
    const float* Ew    = (const float*)d_in[14];
    const float* Eb    = (const float*)d_in[15];

    // ---- workspace carve-out ---------------------------------------------
    char* base = (char*)d_ws;
    size_t off = 0;
    auto carve = [&](size_t bytes) -> void* {
        void* p = base + off;
        off += (bytes + 255) & ~(size_t)255;
        return p;
    };
    float*          xf    = (float*)carve((size_t)Tok * Cd * 4);
    unsigned short* xb    = (unsigned short*)carve((size_t)Tok * Cd * 2);
    unsigned short* Qh    = (unsigned short*)carve((size_t)Tok * Cd * 2);
    unsigned short* Kh    = (unsigned short*)carve((size_t)Tok * Cd * 2);
    unsigned short* Vh    = (unsigned short*)carve((size_t)Tok * Cd * 2);
    unsigned short* KpT   = (unsigned short*)carve((size_t)Bb * Hh * Kdp * Dh * 2);
    unsigned short* Vph   = (unsigned short*)carve((size_t)Bb * Hh * Kdp * Dh * 2);
    float*          Sbuf  = (float*)carve((size_t)Bb * Hh * Ns * Kdp * 4);
    unsigned short* Pbuf  = (unsigned short*)carve((size_t)Bb * Hh * Ns * Kdp * 2);
    unsigned short* oh    = (unsigned short*)carve((size_t)Tok * Cd * 2);
    float*          delta = (float*)carve((size_t)Tok * Cd * 4);
    unsigned short* hbuf  = (unsigned short*)carve((size_t)Tok * Ffd * 2);
    unsigned short* Wq_h  = (unsigned short*)carve((size_t)Ll * Hh * Dh * Cd * 2);
    unsigned short* Wk_h  = (unsigned short*)carve((size_t)Ll * Hh * Dh * Cd * 2);
    unsigned short* Wv_h  = (unsigned short*)carve((size_t)Ll * Hh * Dh * Cd * 2);
    unsigned short* Wo_h  = (unsigned short*)carve((size_t)Ll * Cd * Hh * Dh * 2);
    unsigned short* W1_h  = (unsigned short*)carve((size_t)Ll * Ffd * Cd * 2);
    unsigned short* W2_h  = (unsigned short*)carve((size_t)Ll * Cd * Ffd * 2);
    unsigned short* Ew_h  = (unsigned short*)carve((size_t)Kdp * Ns * 2);

    // ---- one-time bf16 conversions ---------------------------------------
    auto cvt = [&](const float* src, unsigned short* dst, long n) {
        cvt_bf16<<<(unsigned)((n + 255) / 256), 256, 0, stream>>>(src, dst, n);
    };
    copy_cvt<<<(unsigned)(((long)Tok * Cd + 255) / 256), 256, 0, stream>>>(
        x, xf, xb, (long)Tok * Cd);
    cvt(Wq, Wq_h, (long)Ll * Hh * Dh * Cd);
    cvt(Wk, Wk_h, (long)Ll * Hh * Dh * Cd);
    cvt(Wv, Wv_h, (long)Ll * Hh * Dh * Cd);
    cvt(Wo, Wo_h, (long)Ll * Cd * Hh * Dh);
    cvt(W1, W1_h, (long)Ll * Ffd * Cd);
    cvt(W2, W2_h, (long)Ll * Cd * Ffd);
    cvt(Ew, Ew_h, (long)Kdp * Ns);

    // ---- layers ----------------------------------------------------------
    for (int l = 0; l < Ll; ++l) {
        const unsigned short* wq = Wq_h + (size_t)l * Hh * Dh * Cd;
        const unsigned short* wk = Wk_h + (size_t)l * Hh * Dh * Cd;
        const unsigned short* wv = Wv_h + (size_t)l * Hh * Dh * Cd;
        const unsigned short* wo = Wo_h + (size_t)l * Cd * Hh * Dh;
        const unsigned short* w1 = W1_h + (size_t)l * Ffd * Cd;
        const unsigned short* w2 = W2_h + (size_t)l * Cd * Ffd;
        const float* bo_l = bo + (size_t)l * Cd;
        const float* b1_l = b1 + (size_t)l * Ffd;
        const float* b2_l = b2 + (size_t)l * Cd;

        // Q/K/V = xb @ W^T    [16384,512] x [512,512]^T -> bf16 [b,n,h,d]
        launch_gemm(stream, true, xb, wq, nullptr, Qh, nullptr,
                    Tok, Hh * Dh, Cd, Cd, Cd, Hh * Dh,
                    1, 1, 0, 0, 0, 0, 0, 0, 1.0f, 0, 0);
        launch_gemm(stream, true, xb, wk, nullptr, Kh, nullptr,
                    Tok, Hh * Dh, Cd, Cd, Cd, Hh * Dh,
                    1, 1, 0, 0, 0, 0, 0, 0, 1.0f, 0, 0);
        launch_gemm(stream, true, xb, wv, nullptr, Vh, nullptr,
                    Tok, Hh * Dh, Cd, Cd, Cd, Hh * Dh,
                    1, 1, 0, 0, 0, 0, 0, 0, 1.0f, 0, 0);

        // Kp^T[b,h,kd,d] = Ew[kd,n] @ K[b,n,h,d]  (+Eb per-row), batch b*h
        launch_gemm(stream, false, Ew_h, Kh, nullptr, KpT, Eb,
                    Kdp, Dh, Ns, Ns, Cd, Dh,
                    Bb * Hh, Hh, 0, 0, (long)Ns * Cd, (long)Dh,
                    (long)Hh * Kdp * Dh, (long)Kdp * Dh, 1.0f, 2, 0);
        // Vp[b,h,kd,d] = Ew[kd,n] @ V[b,n,h,d]  (+Eb per-row)
        launch_gemm(stream, false, Ew_h, Vh, nullptr, Vph, Eb,
                    Kdp, Dh, Ns, Ns, Cd, Dh,
                    Bb * Hh, Hh, 0, 0, (long)Ns * Cd, (long)Dh,
                    (long)Hh * Kdp * Dh, (long)Kdp * Dh, 1.0f, 2, 0);

        // S[b,h,n,kd] = Q[b,n,h,d] @ Kp^T[b,h,kd,d]^T  (f32)
        launch_gemm(stream, true, Qh, KpT, Sbuf, nullptr, nullptr,
                    Ns, Kdp, Dh, Cd, Dh, Kdp,
                    Bb * Hh, Hh, (long)Ns * Cd, (long)Dh,
                    (long)Hh * Kdp * Dh, (long)Kdp * Dh,
                    (long)Hh * Ns * Kdp, (long)Ns * Kdp, 1.0f, 0, 0);

        // P = softmax(S * 1/sqrt(64))  -> bf16
        softmax256<<<(unsigned)(Bb * Hh * Ns), 256, 0, stream>>>(
            Sbuf, Pbuf, 0.125f);

        // o[b,n,h,d] = P[b,h,n,kd] @ Vp[b,h,kd,d]  -> bf16 concat-heads
        launch_gemm(stream, false, Pbuf, Vph, nullptr, oh, nullptr,
                    Ns, Dh, Kdp, Kdp, Dh, Cd,
                    Bb * Hh, Hh, (long)Hh * Ns * Kdp, (long)Ns * Kdp,
                    (long)Hh * Kdp * Dh, (long)Kdp * Dh,
                    (long)Ns * Cd, (long)Dh, 1.0f, 0, 0);

        // delta = o @ Wo^T + bo  (f32)
        launch_gemm(stream, true, oh, wo, delta, nullptr, bo_l,
                    Tok, Cd, Hh * Dh, Cd, Hh * Dh, Cd,
                    1, 1, 0, 0, 0, 0, 0, 0, 1.0f, 1, 0);

        // x = LN(x + delta)
        residual_ln512<<<(unsigned)Tok, 256, 0, stream>>>(
            xf, delta, ln1_g + (size_t)l * Cd, ln1_b + (size_t)l * Cd, xf, xb);

        // h = gelu(xb @ W1^T + b1)  -> bf16 directly (fused epilogue)
        launch_gemm(stream, true, xb, w1, nullptr, hbuf, b1_l,
                    Tok, Ffd, Cd, Cd, Cd, Ffd,
                    1, 1, 0, 0, 0, 0, 0, 0, 1.0f, 1, 1);
        // delta = h @ W2^T + b2  (f32)
        launch_gemm(stream, true, hbuf, w2, delta, nullptr, b2_l,
                    Tok, Cd, Ffd, Ffd, Ffd, Cd,
                    1, 1, 0, 0, 0, 0, 0, 0, 1.0f, 1, 0);

        // x = LN(x + delta); final layer writes straight to d_out
        float*          xo  = (l == Ll - 1) ? (float*)d_out : xf;
        unsigned short* xbo = (l == Ll - 1) ? nullptr : xb;
        residual_ln512<<<(unsigned)Tok, 256, 0, stream>>>(
            xf, delta, ln2_g + (size_t)l * Cd, ln2_b + (size_t)l * Cd, xo, xbo);
    }
}
